// LightplaneMLPSplatter_31353261261021
// MI455X (gfx1250) — matrix operand order repaired and checked by reading, not compiled, not run on hardware
//
#include <hip/hip_runtime.h>
#include <hip/hip_bf16.h>

typedef __attribute__((ext_vector_type(16))) _Float16 v16h;
typedef __attribute__((ext_vector_type(8)))  _Float16 v8h;
typedef __attribute__((ext_vector_type(8)))  float    v8f;

#define N_RAYS      16384
#define NUM_SAMPLES 96
#define M_TOTAL     (N_RAYS * NUM_SAMPLES)   // 1,572,864 sample points
#define NTILES      (M_TOTAL / 16)           // 98,304 WMMA tiles (16 points each)
#define GR          128
#define C_IN        32
#define C_OUT       16

// ---------------------------------------------------------------- zero output
__global__ void zero_f4(float4* __restrict__ p, int n4) {
  int i = blockIdx.x * blockDim.x + threadIdx.x;
  int stride = gridDim.x * blockDim.x;
  float4 z; z.x = z.y = z.z = z.w = 0.0f;
  for (int j = i; j < n4; j += stride) p[j] = z;
}

// ---------------------------------------------------------------- main kernel
// One wave32 processes 16 sample points per tile iteration.
// Lane L: point p = L&15, half = L>>4.
//  A-layout (16-bit, 16x32): lane half 0 holds channels 0-7 & 16-23 of its row,
//  half 1 holds channels 8-15 & 24-31 -> gather directly into A registers.
__launch_bounds__(256)
__global__ void lightplane_splat(const float* __restrict__ origins,
                                 const float* __restrict__ directions,
                                 const float* __restrict__ nearv,
                                 const float* __restrict__ farv,
                                 const float* __restrict__ enc,
                                 const int*   __restrict__ grid_idx,
                                 const float* __restrict__ grid,   // [B*128^3][32]
                                 const float* __restrict__ W0,     // [32][32]
                                 const float* __restrict__ B0,     // [32]
                                 const float* __restrict__ W1,     // [32][16]
                                 const float* __restrict__ B1,     // [16]
                                 float* __restrict__ out)          // [B*128^3][16]
{
  __shared__ __align__(16) _Float16 hbuf[8][16 * 32];  // per-wave h relayout
  __shared__ __align__(16) float    obuf[8][16 * 16];  // per-wave o relayout

  const int lane      = threadIdx.x & 31;
  const int waveInBlk = threadIdx.x >> 5;
  const int waveId    = blockIdx.x * (blockDim.x >> 5) + waveInBlk;
  const int numWaves  = gridDim.x * (blockDim.x >> 5);

  const int half  = lane >> 4;
  const int n     = lane & 15;        // point index in tile / matrix column
  const int base0 = half * 8;         // first contiguous channel run
  const int base1 = 16 + half * 8;    // second contiguous channel run

  // ---- load weights into B-layout registers (lane = col N; half splits K)
  v16h bw0a, bw0b, bw1;
  #pragma unroll
  for (int e = 0; e < 16; ++e) {
    const int k = (e < 8) ? (base0 + e) : (base1 + e - 8); // B-layout K index
    bw0a[e] = (_Float16)W0[k * 32 + n];                // W0 cols 0..15
    bw0b[e] = (_Float16)W0[k * 32 + 16 + n];           // W0 cols 16..31
    bw1[e]  = (_Float16)W1[k * 16 + n];                // W1 cols 0..15
  }
  const float bias0a = B0[n];
  const float bias0b = B0[16 + n];
  const float bias1  = B1[n];

  _Float16* const hb = hbuf[waveInBlk];
  float*    const ob = obuf[waveInBlk];

  for (int tile = waveId; tile < NTILES; tile += numWaves) {
    // ---------------- per-point geometry (both halves compute point n)
    const int P   = tile * 16 + n;
    const int ray = P / NUM_SAMPLES;
    const int s   = P - ray * NUM_SAMPLES;
    const float tn = nearv[ray], tf = farv[ray];
    const float tv = tn + (tf - tn) * ((float)s * (1.0f / (float)(NUM_SAMPLES - 1)));
    const float px = origins[ray * 3 + 0] + tv * directions[ray * 3 + 0];
    const float py = origins[ray * 3 + 1] + tv * directions[ray * 3 + 1];
    const float pz = origins[ray * 3 + 2] + tv * directions[ray * 3 + 2];
    const int   b  = grid_idx[ray];

    // align_corners=False trilinear setup (same for in/out grids: 128^3)
    const float cx = ((px + 1.0f) * (float)GR - 1.0f) * 0.5f;
    const float cy = ((py + 1.0f) * (float)GR - 1.0f) * 0.5f;
    const float cz = ((pz + 1.0f) * (float)GR - 1.0f) * 0.5f;
    const float fxf = floorf(cx), fyf = floorf(cy), fzf = floorf(cz);
    const int   x0 = (int)fxf, y0 = (int)fyf, z0 = (int)fzf;
    const float fx = cx - fxf, fy = cy - fyf, fz = cz - fzf;

    int   cidx[8];
    float cw[8];
    #pragma unroll
    for (int c = 0; c < 8; ++c) {
      const int dz = (c >> 2) & 1, dy = (c >> 1) & 1, dx = c & 1;
      const int ix = min(max(x0 + dx, 0), GR - 1);
      const int iy = min(max(y0 + dy, 0), GR - 1);
      const int iz = min(max(z0 + dz, 0), GR - 1);
      cw[c]   = (dx ? fx : 1.0f - fx) * (dy ? fy : 1.0f - fy) * (dz ? fz : 1.0f - fz);
      cidx[c] = ((b * GR + iz) * GR + iy) * GR + ix;
    }

    // ---------------- trilinear gather: this lane's 16 A-layout channels
    float f[16];
    #pragma unroll
    for (int e = 0; e < 16; ++e) f[e] = 0.0f;
    #pragma unroll
    for (int c = 0; c < 8; ++c) {
      const float* gp = grid + (size_t)cidx[c] * C_IN;
      const float4 a0 = *(const float4*)(gp + base0);
      const float4 a1 = *(const float4*)(gp + base0 + 4);
      const float4 a2 = *(const float4*)(gp + base1);
      const float4 a3 = *(const float4*)(gp + base1 + 4);
      const float  w  = cw[c];
      f[0]  += w * a0.x; f[1]  += w * a0.y; f[2]  += w * a0.z; f[3]  += w * a0.w;
      f[4]  += w * a1.x; f[5]  += w * a1.y; f[6]  += w * a1.z; f[7]  += w * a1.w;
      f[8]  += w * a2.x; f[9]  += w * a2.y; f[10] += w * a2.z; f[11] += w * a2.w;
      f[12] += w * a3.x; f[13] += w * a3.y; f[14] += w * a3.z; f[15] += w * a3.w;
    }
    {
      const float* ep = enc + (size_t)ray * C_IN;
      const float4 e0 = *(const float4*)(ep + base0);
      const float4 e1 = *(const float4*)(ep + base0 + 4);
      const float4 e2 = *(const float4*)(ep + base1);
      const float4 e3 = *(const float4*)(ep + base1 + 4);
      f[0]  += e0.x; f[1]  += e0.y; f[2]  += e0.z; f[3]  += e0.w;
      f[4]  += e1.x; f[5]  += e1.y; f[6]  += e1.z; f[7]  += e1.w;
      f[8]  += e2.x; f[9]  += e2.y; f[10] += e2.z; f[11] += e2.w;
      f[12] += e3.x; f[13] += e3.y; f[14] += e3.z; f[15] += e3.w;
    }
    v16h a;
    #pragma unroll
    for (int e = 0; e < 16; ++e) a[e] = (_Float16)f[e];

    // ---------------- layer 1: h[16x32] = A[16x32] x W0[32x32]  (2 WMMAs)
    v8f h0 = {}, h1 = {};
    h0 = __builtin_amdgcn_wmma_f32_16x16x32_f16(false, a, false, bw0a,
                                                (short)0, h0, false, false);
    h1 = __builtin_amdgcn_wmma_f32_16x16x32_f16(false, a, false, bw0b,
                                                (short)0, h1, false, false);

    // bias + relu, relayout C/D -> A via per-wave LDS (f16)
    #pragma unroll
    for (int r = 0; r < 8; ++r) {
      const int row = r + 8 * half;
      float v0 = h0[r] + bias0a; v0 = v0 > 0.0f ? v0 : 0.0f;
      float v1 = h1[r] + bias0b; v1 = v1 > 0.0f ? v1 : 0.0f;
      hb[row * 32 + n]      = (_Float16)v0;
      hb[row * 32 + 16 + n] = (_Float16)v1;
    }
    asm volatile("s_wait_dscnt 0" ::: "memory");   // intra-wave LDS RAW

    v16h a2;
    {
      const _Float16* hp = hb + n * 32;            // A row = this lane's point
      const v8h lo = *(const v8h*)(hp + base0);
      const v8h hi = *(const v8h*)(hp + base1);
      a2 = __builtin_shufflevector(lo, hi, 0, 1, 2, 3, 4, 5, 6, 7,
                                   8, 9, 10, 11, 12, 13, 14, 15);
    }

    // ---------------- layer 2: o[16x16] = h[16x32] x W1[32x16]  (1 WMMA)
    v8f o = {};
    o = __builtin_amdgcn_wmma_f32_16x16x32_f16(false, a2, false, bw1,
                                               (short)0, o, false, false);

    // relayout o through LDS so each lane scatters its own point's channels
    #pragma unroll
    for (int r = 0; r < 8; ++r)
      ob[(r + 8 * half) * 16 + n] = o[r] + bias1;
    asm volatile("s_wait_dscnt 0" ::: "memory");

    float ov[16];
    #pragma unroll
    for (int j = 0; j < 4; ++j) {
      const float4 t4 = *(const float4*)(ob + n * 16 + j * 4);
      ov[4 * j + 0] = t4.x; ov[4 * j + 1] = t4.y;
      ov[4 * j + 2] = t4.z; ov[4 * j + 3] = t4.w;
    }

    // ---------------- trilinear scatter: halves split the 8 corners 4/4
    #pragma unroll
    for (int cc = 0; cc < 4; ++cc) {
      const int   c  = half * 4 + cc;
      const float w  = cw[c];
      float*      op = out + (size_t)cidx[c] * C_OUT;
      #pragma unroll
      for (int ch = 0; ch < 16; ++ch)
        unsafeAtomicAdd(op + ch, w * ov[ch]);
    }
  }
}

// ---------------------------------------------------------------- launcher
extern "C" void kernel_launch(void* const* d_in, const int* in_sizes, int n_in,
                              void* d_out, int out_size, void* d_ws, size_t ws_size,
                              hipStream_t stream) {
  const float* origins    = (const float*)d_in[0];
  const float* directions = (const float*)d_in[1];
  const float* nearv      = (const float*)d_in[2];
  const float* farv       = (const float*)d_in[3];
  const float* enc        = (const float*)d_in[4];
  const int*   grid_idx   = (const int*)  d_in[5];
  const float* input_grid = (const float*)d_in[6];
  const float* W0         = (const float*)d_in[7];
  const float* B0         = (const float*)d_in[8];
  const float* W1         = (const float*)d_in[9];
  const float* B1         = (const float*)d_in[10];
  float*       out        = (float*)d_out;

  // zero the 128^3 x 16 output grid (atomically accumulated into)
  const int n4 = out_size / 4;
  zero_f4<<<4096, 256, 0, stream>>>((float4*)out, n4);

  // 256 thr = 8 waves/block; 1024 blocks -> 8192 waves, 12 tiles each
  lightplane_splat<<<1024, 256, 0, stream>>>(origins, directions, nearv, farv,
                                             enc, grid_idx, input_grid,
                                             W0, B0, W1, B1, out);
}
